// LorentzGIN_80607946211343
// MI455X (gfx1250) — matrix operand order repaired and checked by reading, not compiled, not run on hardware
//
#include <hip/hip_runtime.h>

typedef __attribute__((ext_vector_type(16))) __bf16 v16bf;
typedef __attribute__((ext_vector_type(8)))  float  v8f;
typedef __attribute__((ext_vector_type(4)))  unsigned v4u;
typedef __attribute__((ext_vector_type(8)))  int    v8i;
typedef __attribute__((ext_vector_type(4)))  int    v4i;

#define NN 16384
#define DD 128
#define HH 512
#define BTSTRIDE 40   /* LDS B-tile row stride in halfs: 16 dwords + 4 pad dwords */

#if defined(__has_builtin) && __has_builtin(__builtin_amdgcn_sched_barrier)
#define SCHED_FENCE() __builtin_amdgcn_sched_barrier(0)
#else
#define SCHED_FENCE() ((void)0)
#endif

__device__ __forceinline__ unsigned bfbits(float f) {
    unsigned u = __builtin_bit_cast(unsigned, f);
    return (u + 0x7FFFu + ((u >> 16) & 1u)) >> 16;   // RNE
}
__device__ __forceinline__ __bf16 f2bf(float f) {
    unsigned short h = (unsigned short)bfbits(f);
    return __builtin_bit_cast(__bf16, h);
}
__device__ __forceinline__ float wsum(float v) {     // wave32 all-reduce
    for (int o = 16; o > 0; o >>= 1) v += __shfl_xor(v, o, 32);
    return v;
}
__device__ __forceinline__ float acosh_f(float z) {
    z = fmaxf(z, 1.0f + 1e-7f);
    return logf(z + sqrtf(z * z - 1.0f));
}

// ------------------------------------------- transpose-convert f32 -> bf16
__global__ __launch_bounds__(256) void k_cvtT(const float* __restrict__ s,
                                              __bf16* __restrict__ d, int R, int C) {
    int i = blockIdx.x * 256 + threadIdx.x;
    if (i < R * C) {
        int r = i / C, c = i - r * C;
        d[(size_t)c * R + r] = f2bf(s[i]);
    }
}

// ------------------------------- lh = log_map(exp_map(x)); emit f32 + bf16^T
__global__ __launch_bounds__(256) void k_logexp(const float* __restrict__ x,
                                                float* __restrict__ lhf,
                                                __bf16* __restrict__ lhbT) {
    int w = threadIdx.x >> 5, lane = threadIdx.x & 31;
    int row = blockIdx.x * 8 + w;
    const float* xr = x + (size_t)row * DD;
    float u0 = xr[lane], u1 = xr[lane + 32], u2 = xr[lane + 64], u3 = xr[lane + 96];
    float s = u1 * u1 + u2 * u2 + u3 * u3 + (lane ? u0 * u0 : 0.0f);
    s = wsum(s);
    float n1 = fmaxf(sqrtf(s), 1e-7f);
    float sh = sinhf(n1), ch = coshf(n1), iv = sh / n1;
    float h0 = lane ? iv * u0 : ch;
    float h1 = iv * u1, h2 = iv * u2, h3 = iv * u3;
    float s2 = h1 * h1 + h2 * h2 + h3 * h3 + (lane ? h0 * h0 : 0.0f);
    s2 = wsum(s2);
    float n2 = fmaxf(sqrtf(s2), 1e-7f);
    float sc = acosh_f(ch) / n2;
    float l0 = lane ? sc * h0 : 0.0f;
    float l1 = sc * h1, l2 = sc * h2, l3 = sc * h3;
    size_t b = (size_t)row * DD;
    lhf[b + lane] = l0;       lhf[b + lane + 32] = l1;
    lhf[b + lane + 64] = l2;  lhf[b + lane + 96] = l3;
    lhbT[(size_t)(lane)       * NN + row] = f2bf(l0);
    lhbT[(size_t)(lane + 32)  * NN + row] = f2bf(l1);
    lhbT[(size_t)(lane + 64)  * NN + row] = f2bf(l2);
    lhbT[(size_t)(lane + 96)  * NN + row] = f2bf(l3);
}

// ---------------------- TDM: 128x32 bf16 tile from lhbT (D x N) -> LDS
__device__ __forceinline__ void tdm_load_B(const __bf16* lhbT, unsigned lds_off, int k0) {
    unsigned long long ga =
        (unsigned long long)(size_t)lhbT + (unsigned long long)k0 * 2;
    v4u g0 = { 1u, lds_off, (unsigned)ga,
               (unsigned)((ga >> 32) & 0x01FFFFFFu) | 0x80000000u };
    v8i g1 = { 0x06D10000,           // data_size=1(2B)|pad_en|pad_int=3|pad_amt=3
               (int)(16384u << 16),  // tensor_dim0 = NN (low16 << 16)
               (int)(128u << 16),    // tensor_dim1 = 128
               (int)(32u << 16),     // tile_dim0 = 32
               128,                  // tile_dim1 = 128
               16384, 0, 0 };        // tensor_dim0_stride = NN
    v4i gz = { 0, 0, 0, 0 };
#if __clang_major__ >= 23
    v8i gz8 = { 0, 0, 0, 0, 0, 0, 0, 0 };
    __builtin_amdgcn_tensor_load_to_lds(g0, g1, gz, gz, gz8, 0);
#else
    __builtin_amdgcn_tensor_load_to_lds(g0, g1, gz, gz, 0);
#endif
}

// -------------------------------------------------- fragment assembly
__device__ __forceinline__ void load_fragA1(const __bf16 (*As)[40], int mrow, int half,
                                            v16bf& a) {
    const __bf16* p = &As[mrow][half * 16];
    #pragma unroll
    for (int t = 0; t < 8; ++t) { a[t] = p[t]; a[8 + t] = p[8 + t]; }
}
__device__ __forceinline__ void load_fragB(const __bf16* Bt, int col, int half, v16bf& b) {
    const __bf16* pb = &Bt[(size_t)col * BTSTRIDE + half * 16];
    #pragma unroll
    for (int j = 0; j < 16; ++j) b[j] = pb[j];   // 32B contiguous -> 2x ds_load_b128
}

// -------------------------------------------------- agg GEMM + hyperbolic epilogue
// 64 rows x 128 cols per block; 8 waves, each wave: 16-col slice, 4 WMMAs/K-step.
__global__ __launch_bounds__(256) void k_agg(const float* __restrict__ adj,
                                             const __bf16* __restrict__ lhbT,
                                             const float* __restrict__ lhf,
                                             __bf16* __restrict__ outb) {
    __shared__ __align__(16) __bf16 As[2][64][40];           // adj tile (bf16)
    __shared__ __align__(16) __bf16 Bs[2][128 * BTSTRIDE];   // lh^T tile via TDM
    __shared__ __align__(16) float  Ct[64][132];
    int tid = threadIdx.x, lane = tid & 31, w = tid >> 5;
    int row0 = blockIdx.x * 64;
    int cb = w * 16, half = lane >> 4, l15 = lane & 15;
    int ar = tid >> 2, ac = (tid & 3) * 8;    // A staging: 8 floats/thread
    v8f acc[4] = {};

    // ---- prologue: stage tile 0
    if (w == 0) tdm_load_B(lhbT, (unsigned)(size_t)&Bs[0][0], 0);
    {
        const float4* src = (const float4*)&adj[(size_t)(row0 + ar) * NN + ac];
        float4 f0 = src[0], f1 = src[1];
        uint4 pk = make_uint4(bfbits(f0.x) | (bfbits(f0.y) << 16),
                              bfbits(f0.z) | (bfbits(f0.w) << 16),
                              bfbits(f1.x) | (bfbits(f1.y) << 16),
                              bfbits(f1.z) | (bfbits(f1.w) << 16));
        *(uint4*)&As[0][ar][ac] = pk;
    }
    if (w == 0) __builtin_amdgcn_s_wait_tensorcnt(0);
    __syncthreads();

    for (int i = 0; i < NN / 32; ++i) {
        int cur = i & 1, nxt = cur ^ 1;
        int k0n = (i + 1) * 32;
        bool has_next = k0n < NN;
        // issue next-tile fetches early: TDM for B, global_load_b128 x2 for A
        if (has_next && w == 0) tdm_load_B(lhbT, (unsigned)(size_t)&Bs[nxt][0], k0n);
        float4 f0, f1;
        if (has_next) {
            const float4* src = (const float4*)&adj[(size_t)(row0 + ar) * NN + k0n + ac];
            f0 = src[0]; f1 = src[1];
            if (tid < 64 && k0n + 32 < NN)   // prefetch next-next tile into SE/GL2
                __builtin_prefetch(&adj[(size_t)(row0 + tid) * NN + k0n + 32], 0, 3);
        }
        SCHED_FENCE();   // keep the global-load issue above, and nothing below it
        // compute on current tile (depends only on LDS -> waits only on dscnt)
        v16bf bf;
        load_fragB(Bs[cur], cb + l15, half, bf);
        #pragma unroll
        for (int mt = 0; mt < 4; ++mt) {
            v16bf a;
            load_fragA1(As[cur], mt * 16 + l15, half, a);
            acc[mt] = __builtin_amdgcn_wmma_f32_16x16x32_bf16(false, a, false, bf,
                                                              (short)0, acc[mt],
                                                              false, false);
        }
        SCHED_FENCE();   // pack/store (loadcnt wait) must stay below the WMMAs
        // commit next A tile to LDS
        if (has_next) {
            uint4 pk = make_uint4(bfbits(f0.x) | (bfbits(f0.y) << 16),
                                  bfbits(f0.z) | (bfbits(f0.w) << 16),
                                  bfbits(f1.x) | (bfbits(f1.y) << 16),
                                  bfbits(f1.z) | (bfbits(f1.w) << 16));
            *(uint4*)&As[nxt][ar][ac] = pk;
        }
        if (has_next && w == 0) __builtin_amdgcn_s_wait_tensorcnt(0);
        __syncthreads();
    }
    #pragma unroll
    for (int mt = 0; mt < 4; ++mt)
        #pragma unroll
        for (int v = 0; v < 8; ++v)
            Ct[mt * 16 + v + 8 * half][cb + l15] = acc[mt][v];
    __syncthreads();

    // epilogue: exp_map -> log_map -> +lh -> exp_map; wave w: rows 8w..8w+7
    for (int rr = 0; rr < 8; ++rr) {
        int r = w * 8 + rr, grow = row0 + r;
        float u[4], vv[4];
        float s = 0.f;
        #pragma unroll
        for (int j = 0; j < 4; ++j) {
            int c = lane + 32 * j; u[j] = Ct[r][c];
            if (c) s += u[j] * u[j];
        }
        s = wsum(s);
        float n1 = fmaxf(sqrtf(s), 1e-7f);
        float sh1 = sinhf(n1), ch1 = coshf(n1), iv1 = sh1 / n1;
        float s2 = 0.f;
        #pragma unroll
        for (int j = 0; j < 4; ++j) {
            int c = lane + 32 * j; u[j] = c ? iv1 * u[j] : ch1;
            if (c) s2 += u[j] * u[j];
        }
        s2 = wsum(s2);
        float n2 = fmaxf(sqrtf(s2), 1e-7f);
        float sc = acosh_f(ch1) / n2;
        float s3 = 0.f;
        #pragma unroll
        for (int j = 0; j < 4; ++j) {
            int c = lane + 32 * j;
            vv[j] = (c ? sc * u[j] : 0.f) + lhf[(size_t)grow * DD + c];
            if (c) s3 += vv[j] * vv[j];
        }
        s3 = wsum(s3);
        float n3 = fmaxf(sqrtf(s3), 1e-7f);
        float sh3 = sinhf(n3), ch3 = coshf(n3), iv3 = sh3 / n3;
        #pragma unroll
        for (int j = 0; j < 4; ++j) {
            int c = lane + 32 * j;
            outb[(size_t)grow * DD + c] = f2bf(c ? iv3 * vv[j] : ch3);
        }
    }
}

// ---------------------------------- MLP staging: A row-major, B^T contiguous
__device__ __forceinline__ void stage_mlp(const __bf16* __restrict__ Asrc, int astride,
                                          const __bf16* __restrict__ BTsrc, int btstride,
                                          __bf16 (*As)[40], __bf16 (*Bt)[BTSTRIDE],
                                          int row0, int k0, int col0, int tid) {
    if (tid < 128) {                       // A: 32x32 halfs, 8 halfs/thread
        int r = tid >> 2, c = (tid & 3) * 8;
        *(uint4*)&As[r][c] = *(const uint4*)&Asrc[(size_t)(row0 + r) * astride + k0 + c];
    }
    {                                      // B^T: 128 cols x 32 halfs, 16/thread
        int r = tid >> 1, c = (tid & 1) * 16;
        const uint4* s = (const uint4*)&BTsrc[(size_t)(col0 + r) * btstride + k0 + c];
        uint4* d = (uint4*)&Bt[r][c];
        d[0] = s[0]; d[1] = s[1];
    }
}

// -------------------------------------------------- hh = relu(out @ W1 + b1)
__global__ __launch_bounds__(256) void k_mlp1(const __bf16* __restrict__ outb,
                                              const __bf16* __restrict__ W1bT,
                                              const float* __restrict__ b1,
                                              __bf16* __restrict__ hhb) {
    __shared__ __align__(16) __bf16 As[32][40];
    __shared__ __align__(16) __bf16 Bt[128][BTSTRIDE];
    int tid = threadIdx.x, lane = tid & 31, w = tid >> 5;
    int row0 = blockIdx.x * 32, col0 = blockIdx.y * 128;
    int cb = w * 16, half = lane >> 4, l15 = lane & 15;
    v8f acc0 = {}, acc1 = {};
    for (int k0 = 0; k0 < DD; k0 += 32) {
        stage_mlp(outb, DD, W1bT, DD, As, Bt, row0, k0, col0, tid);
        __syncthreads();
        v16bf a0, a1, bf;
        load_fragB(&Bt[0][0], cb + l15, half, bf);
        load_fragA1(As, l15, half, a0);
        load_fragA1(As, 16 + l15, half, a1);
        acc0 = __builtin_amdgcn_wmma_f32_16x16x32_bf16(false, a0, false, bf,
                                                       (short)0, acc0, false, false);
        acc1 = __builtin_amdgcn_wmma_f32_16x16x32_bf16(false, a1, false, bf,
                                                       (short)0, acc1, false, false);
        __syncthreads();
    }
    int col = col0 + cb + l15;
    float bias = b1[col];
    #pragma unroll
    for (int v = 0; v < 8; ++v) {
        int r0 = row0 + v + 8 * half;
        hhb[(size_t)r0 * HH + col]        = f2bf(fmaxf(acc0[v] + bias, 0.f));
        hhb[(size_t)(r0 + 16) * HH + col] = f2bf(fmaxf(acc1[v] + bias, 0.f));
    }
}

// -------------------------------------------------- t = hh @ W2 + b2 (f32 out)
__global__ __launch_bounds__(256) void k_mlp2(const __bf16* __restrict__ hhb,
                                              const __bf16* __restrict__ W2bT,
                                              const float* __restrict__ b2,
                                              float* __restrict__ out) {
    __shared__ __align__(16) __bf16 As[32][40];
    __shared__ __align__(16) __bf16 Bt[128][BTSTRIDE];
    int tid = threadIdx.x, lane = tid & 31, w = tid >> 5;
    int row0 = blockIdx.x * 32;
    int cb = w * 16, half = lane >> 4, l15 = lane & 15;
    v8f acc0 = {}, acc1 = {};
    for (int k0 = 0; k0 < HH; k0 += 32) {
        stage_mlp(hhb, HH, W2bT, HH, As, Bt, row0, k0, 0, tid);
        __syncthreads();
        v16bf a0, a1, bf;
        load_fragB(&Bt[0][0], cb + l15, half, bf);
        load_fragA1(As, l15, half, a0);
        load_fragA1(As, 16 + l15, half, a1);
        acc0 = __builtin_amdgcn_wmma_f32_16x16x32_bf16(false, a0, false, bf,
                                                       (short)0, acc0, false, false);
        acc1 = __builtin_amdgcn_wmma_f32_16x16x32_bf16(false, a1, false, bf,
                                                       (short)0, acc1, false, false);
        __syncthreads();
    }
    int col = cb + l15;
    float bias = b2[col];
    #pragma unroll
    for (int v = 0; v < 8; ++v) {
        int r0 = row0 + v + 8 * half;
        out[(size_t)r0 * DD + col]        = acc0[v] + bias;
        out[(size_t)(r0 + 16) * DD + col] = acc1[v] + bias;
    }
}

extern "C" void kernel_launch(void* const* d_in, const int* in_sizes, int n_in,
                              void* d_out, int out_size, void* d_ws, size_t ws_size,
                              hipStream_t stream) {
    (void)in_sizes; (void)n_in; (void)out_size; (void)ws_size;
    const float* x   = (const float*)d_in[0];
    const float* adj = (const float*)d_in[1];
    const float* W1  = (const float*)d_in[2];
    const float* b1  = (const float*)d_in[3];
    const float* W2  = (const float*)d_in[4];
    const float* b2  = (const float*)d_in[5];
    float* out = (float*)d_out;

    char* p = (char*)d_ws;
    auto alloc = [&](size_t bytes) -> char* {
        char* r = p; p += (bytes + 255) & ~(size_t)255; return r;
    };
    float*  lhf  = (float*)alloc(sizeof(float) * (size_t)NN * DD);
    __bf16* lhbT = (__bf16*)alloc(2ull * NN * DD);   // D x N (transposed)
    __bf16* outb = (__bf16*)alloc(2ull * NN * DD);   // N x D
    __bf16* hhb  = (__bf16*)alloc(2ull * NN * HH);   // N x H
    __bf16* W1bT = (__bf16*)alloc(2ull * DD * HH);   // H x D (transposed)
    __bf16* W2bT = (__bf16*)alloc(2ull * HH * DD);   // D x H (transposed)

    k_cvtT<<<(DD * HH + 255) / 256, 256, 0, stream>>>(W1, W1bT, DD, HH);
    k_cvtT<<<(HH * DD + 255) / 256, 256, 0, stream>>>(W2, W2bT, HH, DD);
    k_logexp<<<NN / 8, 256, 0, stream>>>(x, lhf, lhbT);
    k_agg<<<NN / 64, 256, 0, stream>>>(adj, lhbT, lhf, outb);
    k_mlp1<<<dim3(NN / 32, HH / 128), 256, 0, stream>>>(outb, W1bT, b1, hhb);
    k_mlp2<<<NN / 32, 256, 0, stream>>>(hhb, W2bT, b2, out);
}
